// BaseLSTMModel_44083544326149
// MI455X (gfx1250) — matrix operand order repaired and checked by reading, not compiled, not run on hardware
//
#include <hip/hip_runtime.h>
#include <hip/hip_bf16.h>

#define N_  8
#define T_  128
#define H_  32
#define W_  32
#define C_  3
#define HID 128
#define CF  32
#define FLAT 32768   // H*W*CF
#define FRAMES 1024  // N_*T_
#define KSPLIT 8     // K-dimension splits for the big GEMM

typedef __attribute__((ext_vector_type(16))) __bf16 v16bf;
typedef __attribute__((ext_vector_type(8)))  __bf16 v8bf;
typedef __attribute__((ext_vector_type(8)))  float  v8f;

__device__ __forceinline__ unsigned short f2bf(float f) {
  unsigned int u = __float_as_uint(f);
  u += 0x7FFFu + ((u >> 16) & 1u);   // round-to-nearest-even
  return (unsigned short)(u >> 16);
}

// ---------------------------------------------------------------------------
// Stage 0: LDS-tiled transpose early_w [FLAT,HID] fp32 -> Bt [HID][FLAT] bf16
//   Both global read and write are coalesced; grid = 1024 k-tiles x 4 n-tiles.
// ---------------------------------------------------------------------------
__global__ __launch_bounds__(256) void prep_bt(
    const float* __restrict__ ew, unsigned short* __restrict__ Bt) {
  __shared__ unsigned short tile[32][33];
  const int tid = threadIdx.x;
  const int k0 = (blockIdx.x >> 2) << 5;
  const int n0 = (blockIdx.x & 3) << 5;
  {
    const int nl = tid & 31, kq = tid >> 5;
    #pragma unroll
    for (int i = 0; i < 4; ++i) {
      int kl = kq + (i << 3);
      tile[kl][nl] = f2bf(ew[(k0 + kl) * HID + n0 + nl]);
    }
  }
  __syncthreads();
  {
    const int kl = tid & 31, nq = tid >> 5;
    #pragma unroll
    for (int i = 0; i < 4; ++i) {
      int nl = nq + (i << 3);
      Bt[(size_t)(n0 + nl) * FLAT + k0 + kl] = tile[kl][nl];
    }
  }
}

// ---------------------------------------------------------------------------
// Stage 1: fused conv(3x3, SAME, implicit-GEMM WMMA) + early MLP GEMM (WMMA)
//   grid = (64 M-tiles of 16 frames) x (KSPLIT splits). 256 threads = 8 waves.
//   Per chunk (16 spatial positions, Kc=512):
//     pf: cooperative global_prefetch of next chunk's Bt slice (overlaps conv)
//     A1: im2col [256 rows x 32(K=27 pad)] bf16 into LDS
//     A2: conv = im2col @ conv_w via 4 WMMAs/wave, +conv_b, bf16 -> sA
//     B : 16 WMMAs/wave of sA[16x512] @ Bt-slice, fp32 accumulate
//   Each split writes a non-atomic fp32 partial [1024x128] (determinism).
// ---------------------------------------------------------------------------
__global__ __launch_bounds__(256) void conv_early_gemm(
    const float* __restrict__ x, const float* __restrict__ conv_w,
    const float* __restrict__ conv_b, const unsigned short* __restrict__ Bt,
    float* __restrict__ partials) {
  __shared__ __align__(16) unsigned short sIM[256 * 32];  // 16KB im2col, bf16
  __shared__ __align__(16) unsigned short sA[16 * 512];   // 16KB conv-out, bf16
  __shared__ __align__(16) unsigned short sWB[32 * 32];   // 2KB conv W [filt][k]
  __shared__ float sCB[32];

  const int tid    = threadIdx.x;
  const int frBase = blockIdx.x * 16;
  const int split  = blockIdx.y;

  for (int i = tid; i < 1024; i += 256) {
    int filt = i >> 5, k = i & 31;
    sWB[i] = (k < 27) ? f2bf(conv_w[k * CF + filt]) : (unsigned short)0;
  }
  if (tid < 32) sCB[tid] = conv_b[tid];

  const int lane   = tid & 31;
  const int wid    = tid >> 5;      // wave id
  const int n0     = wid * 16;      // N-tile of the main GEMM
  const int lrow   = lane & 15;
  const int hiHalf = lane >> 4;

  v8f acc = {0.f, 0.f, 0.f, 0.f, 0.f, 0.f, 0.f, 0.f};

  for (int cc = 0; cc < 64 / KSPLIT; ++cc) {
    const int chunk = split * (64 / KSPLIT) + cc;

    // ---- pf: prefetch next chunk's B slice (128 rows x 1KB) while conv runs
    if (cc + 1 < 64 / KSPLIT) {
      const unsigned short* nb =
          Bt + (size_t)(tid >> 1) * FLAT + ((chunk + 1) << 9) + ((tid & 1) << 8);
      __builtin_prefetch(nb, 0, 3);
      __builtin_prefetch((const char*)nb + 128, 0, 3);
      __builtin_prefetch((const char*)nb + 256, 0, 3);
      __builtin_prefetch((const char*)nb + 384, 0, 3);
    }

    __syncthreads();
    // ---- A1: im2col tile (rows = 16 frames x 16 positions, K = 27 pad 32) --
    for (int i = 0; i < 32; ++i) {
      int e    = tid + (i << 8);       // 0..8191
      int row  = e >> 5;               // 0..255
      int kidx = e & 31;
      int fl = row >> 4, pl = row & 15;
      int p = (chunk << 4) + pl;
      int h = p >> 5, w = p & 31;
      float v = 0.f;
      if (kidx < 27) {
        int c  = kidx % 3;
        int r9 = kidx / 3;
        int dh = r9 / 3, dw = r9 % 3;
        int hh = h + dh - 1, ww = w + dw - 1;
        if (hh >= 0 && hh < H_ && ww >= 0 && ww < W_)
          v = x[(((frBase + fl) * H_ + hh) * W_ + ww) * C_ + c];
      }
      sIM[row * 32 + kidx] = f2bf(v);
    }
    __syncthreads();
    // ---- A2: conv via WMMA (16 M-tiles x 2 N-tiles = 32 jobs, 4/wave) ------
    #pragma unroll
    for (int jj = 0; jj < 4; ++jj) {
      int job = (wid << 2) + jj;
      int mt = job >> 1;
      int nt = job & 1;
      const unsigned short* ap = sIM + ((mt << 4) + lrow) * 32 + (hiHalf ? 8 : 0);
      union { v16bf v; v8bf h2[2]; } A;
      A.h2[0] = *reinterpret_cast<const v8bf*>(ap);
      A.h2[1] = *reinterpret_cast<const v8bf*>(ap + 16);
      int filt = (nt << 4) + lrow;
      union { v16bf v; v8bf h2[2]; } B;
      B.h2[0] = *reinterpret_cast<const v8bf*>(sWB + filt * 32 + (hiHalf ? 8 : 0));
      B.h2[1] = *reinterpret_cast<const v8bf*>(sWB + filt * 32 + (hiHalf ? 8 : 0) + 16);
      v8f cv = {0.f, 0.f, 0.f, 0.f, 0.f, 0.f, 0.f, 0.f};
      cv = __builtin_amdgcn_wmma_f32_16x16x32_bf16(
          false, A.v, false, B.v, (short)0, cv, false, false);
      float cb = sCB[filt];
      #pragma unroll
      for (int r = 0; r < 8; ++r) {
        int rowg = (mt << 4) + r + (hiHalf ? 8 : 0);
        int fl = rowg >> 4, pl = rowg & 15;
        sA[fl * 512 + (pl << 5) + filt] = f2bf(cv[r] + cb);
      }
    }
    __syncthreads();
    // ---- B: main GEMM, 16 WMMAs over this K-chunk --------------------------
    const unsigned short* brow =
        Bt + (size_t)(n0 + lrow) * FLAT + (chunk << 9) + (hiHalf ? 16 : 0);
    const unsigned short* arow = sA + lrow * 512 + (hiHalf ? 16 : 0);
    #pragma unroll 4
    for (int kk0 = 0; kk0 < 512; kk0 += 32) {
      union { v16bf v; v8bf h2[2]; } A;
      A.h2[0] = *reinterpret_cast<const v8bf*>(arow + kk0);
      A.h2[1] = *reinterpret_cast<const v8bf*>(arow + kk0 + 8);
      v16bf B = *reinterpret_cast<const v16bf*>(brow + kk0);
      acc = __builtin_amdgcn_wmma_f32_16x16x32_bf16(
          false, A.v, false, B, (short)0, acc, false, false);
    }
  }
  // ---- store fp32 partial tile (bias/relu applied after reduction) --------
  float* pp = partials + (size_t)split * (FRAMES * HID);
  #pragma unroll
  for (int r = 0; r < 8; ++r) {
    int m = r + (hiHalf ? 8 : 0);
    pp[(frBase + m) * HID + n0 + lrow] = acc[r];
  }
}

// ---------------------------------------------------------------------------
// Stage 1b: sum K-split partials + early bias + relu -> feat [1024,128]
// ---------------------------------------------------------------------------
__global__ __launch_bounds__(256) void reduce_bias_relu(
    const float* __restrict__ partials, const float* __restrict__ eb,
    float* __restrict__ feat) {
  int idx = blockIdx.x * 256 + threadIdx.x;   // 131072
  float a = eb[idx & (HID - 1)];
  #pragma unroll
  for (int s = 0; s < KSPLIT; ++s) a += partials[(size_t)s * (FRAMES * HID) + idx];
  feat[idx] = a > 0.f ? a : 0.f;
}

// ---------------------------------------------------------------------------
// Stage 2: zx = feat[1024,128] @ lstm_wx[128,512] + lstm_b   (fp32 VALU)
// ---------------------------------------------------------------------------
__global__ __launch_bounds__(256) void zx_gemm(
    const float* __restrict__ feat, const float* __restrict__ wx,
    const float* __restrict__ b, float* __restrict__ zx) {
  int idx = blockIdx.x * 256 + threadIdx.x;   // 1024*512
  int fr = idx >> 9;
  int j  = idx & 511;
  const float* fp = feat + fr * HID;
  float a = b[j];
  #pragma unroll 8
  for (int k = 0; k < HID; ++k) a += fp[k] * wx[k * 512 + j];
  zx[idx] = a;
}

// ---------------------------------------------------------------------------
// Stage 3: reset-masked LSTM scan, one block per batch row (rows independent)
// ---------------------------------------------------------------------------
__global__ __launch_bounds__(256) void lstm_scan(
    const float* __restrict__ zx, const int* __restrict__ done,
    const float* __restrict__ wh, float* __restrict__ hs,
    float* __restrict__ out /* c_fin @0, h_fin @1024 */) {
  __shared__ float hC[HID], cC[HID], zS[4 * HID];
  const int tid = threadIdx.x;
  const int n   = blockIdx.x;
  if (tid < HID) { hC[tid] = 0.f; cC[tid] = 0.f; }
  for (int t = 0; t < T_; ++t) {
    const int fr    = n * T_ + t;
    const int dmask = done[fr];
    __syncthreads();
    if (dmask && tid < HID) { hC[tid] = 0.f; cC[tid] = 0.f; }
    __syncthreads();
    float a0 = zx[fr * 512 + tid];
    float a1 = zx[fr * 512 + 256 + tid];
    for (int k = 0; k < HID; ++k) {
      float hk = hC[k];
      a0 += hk * wh[k * 512 + tid];
      a1 += hk * wh[k * 512 + 256 + tid];
    }
    zS[tid] = a0; zS[tid + 256] = a1;
    __syncthreads();
    if (tid < HID) {
      float ig = 1.f / (1.f + __expf(-zS[tid]));
      float fg = 1.f / (1.f + __expf(-zS[HID + tid]));
      float gg = tanhf(zS[2 * HID + tid]);
      float og = 1.f / (1.f + __expf(-zS[3 * HID + tid]));
      float c  = fg * cC[tid] + ig * gg;
      float h  = og * tanhf(c);
      cC[tid] = c; hC[tid] = h;
      hs[fr * HID + tid] = h;
    }
  }
  __syncthreads();
  if (tid < HID) {
    out[n * HID + tid]            = cC[tid];
    out[N_ * HID + n * HID + tid] = hC[tid];
  }
}

// ---------------------------------------------------------------------------
// Stage 4: y = relu(hs[1024,128] @ out_w[128,128] + out_b)   (fp32 VALU)
// ---------------------------------------------------------------------------
__global__ __launch_bounds__(256) void out_mlp(
    const float* __restrict__ hs, const float* __restrict__ ow,
    const float* __restrict__ ob, float* __restrict__ y) {
  int idx = blockIdx.x * 256 + threadIdx.x;   // 1024*128
  int fr = idx >> 7;
  int j  = idx & 127;
  const float* hp = hs + fr * HID;
  float a = ob[j];
  #pragma unroll 8
  for (int k = 0; k < HID; ++k) a += hp[k] * ow[k * HID + j];
  y[idx] = a > 0.f ? a : 0.f;
}

// ---------------------------------------------------------------------------
extern "C" void kernel_launch(void* const* d_in, const int* in_sizes, int n_in,
                              void* d_out, int out_size, void* d_ws, size_t ws_size,
                              hipStream_t stream) {
  (void)in_sizes; (void)n_in; (void)out_size; (void)ws_size;
  const float* x       = (const float*)d_in[0];
  const int*   done    = (const int*)  d_in[1];
  const float* conv_w  = (const float*)d_in[2];
  const float* conv_b  = (const float*)d_in[3];
  const float* early_w = (const float*)d_in[4];
  const float* early_b = (const float*)d_in[5];
  const float* lstm_wx = (const float*)d_in[6];
  const float* lstm_wh = (const float*)d_in[7];
  const float* lstm_b  = (const float*)d_in[8];
  const float* out_w   = (const float*)d_in[9];
  const float* out_b   = (const float*)d_in[10];
  float* out = (float*)d_out;

  char* ws = (char*)d_ws;
  unsigned short* Bt   = (unsigned short*)ws;                  //  8,388,608 B
  float* partials = (float*)(ws + 8388608);                    //  4,194,304 B
  float* feat     = (float*)(ws + 8388608 + 4194304);          //    524,288 B
  float* zx       = (float*)(ws + 8388608 + 4194304 + 524288); //  2,097,152 B
  float* hs       = (float*)(ws + 8388608 + 4194304 + 524288 + 2097152); // 524,288 B

  prep_bt        <<<4096, 256, 0, stream>>>(early_w, Bt);
  conv_early_gemm<<<dim3(64, KSPLIT), 256, 0, stream>>>(x, conv_w, conv_b, Bt, partials);
  reduce_bias_relu<<<512, 256, 0, stream>>>(partials, early_b, feat);
  zx_gemm        <<<2048, 256, 0, stream>>>(feat, lstm_wx, lstm_b, zx);
  lstm_scan      <<<   8, 256, 0, stream>>>(zx, done, lstm_wh, hs, out);
  out_mlp        <<< 512, 256, 0, stream>>>(hs, out_w, out_b, out + 2048);
}
